// QuantizedSpectralConv_82446192214616
// MI455X (gfx1250) — matrix operand order repaired and checked by reading, not compiled, not run on hardware
//
#include <hip/hip_runtime.h>
#include <hip/hip_bf16.h>

// ---------------------------------------------------------------------------
// QuantizedSpectralConv on MI455X (gfx1250):
// pruned-DFT spectral convolution, all heavy math on v_wmma_f32_16x16x32_bf16.
// Twiddle tables staged into LDS via the Tensor Data Mover (TDM).
// B=16, CIN=64, COUT=64, H=W=256, retained modes 64x64 (kx=96..159, ky=32..95)
// ---------------------------------------------------------------------------

typedef __attribute__((ext_vector_type(16))) __bf16 v16bf;
typedef __attribute__((ext_vector_type(8)))  __bf16 v8bf;
typedef __attribute__((ext_vector_type(8)))  float  v8f;
typedef __attribute__((ext_vector_type(4)))  unsigned int v4u;
typedef __attribute__((ext_vector_type(8)))  int v8i;
typedef __attribute__((ext_vector_type(4)))  int v4i;

static constexpr int Bn   = 16;
static constexpr int CIN  = 64;
static constexpr int COUT = 64;
static constexpr int Hh   = 256;
static constexpr int Ww   = 256;
static constexpr int KM   = 64;   // retained modes per axis
static constexpr int KX0  = 96;   // kx slice start (cx - nx = 128-32)
static constexpr int KY0  = 32;   // ky slice start (cy - ny = 64-32)

__device__ __forceinline__ v8f wmma_bf16(v16bf a, v16bf b, v8f c) {
  // D = A(16x32) * B(32x16) + C, f32 accumulate
  return __builtin_amdgcn_wmma_f32_16x16x32_bf16(false, a, false, b, (short)0, c,
                                                 false, false);
}

// ---------------------------------------------------------------------------
// TDM: flat copy of n8 8-byte elements from global to LDS (single-row 2D tile).
// D# per CDNA5 ISA ch.8: group0 = {count=1, lds_addr, global_addr[56:0], type=2}
// group1 = {mask=0, data_size=8B, tensor_dim0=tile_dim0=n8, tile_dim1=1}
// groups 2/3 (and trailing group) zero: <=2D tensor, unused dims.
// ---------------------------------------------------------------------------
__device__ __forceinline__ unsigned lds_addr_of(const void* p) {
  return (unsigned)(unsigned long long)p;   // low 32 bits of LDS aperture addr
}

__device__ __forceinline__ void tdm_load_1d(unsigned lds_off, const void* gptr,
                                            unsigned n8) {
  unsigned long long ga = (unsigned long long)gptr;
  v4u g0;
  g0[0] = 1u;                                          // count=1, user mode
  g0[1] = lds_off;                                     // lds_addr (bytes)
  g0[2] = (unsigned)(ga & 0xFFFFFFFFu);                // global_addr[31:0]
  g0[3] = (unsigned)((ga >> 32) & 0x01FFFFFFu)         // global_addr[56:32]
        | (2u << 30);                                  // type=2 ("image")
  v8i g1;
  g1[0] = (int)(3u << 16);                             // wg_mask=0, data_size=8B
  g1[1] = (int)((n8 & 0xFFFFu) << 16);                 // tensor_dim0[15:0]
  g1[2] = (int)(((n8 >> 16) & 0xFFFFu) | (1u << 16));  // dim0[31:16], tensor_dim1=1
  g1[3] = (int)((n8 & 0xFFFFu) << 16);                 // tile_dim0 (=n8, <=65535)
  g1[4] = 1;                                           // tile_dim1=1, tile_dim2=0
  g1[5] = (int)n8;                                     // tensor_dim0_stride[31:0]
  g1[6] = 0;
  g1[7] = 0;
  v4i z4 = {0, 0, 0, 0};
  v8i z8 = {0, 0, 0, 0, 0, 0, 0, 0};
  __builtin_amdgcn_tensor_load_to_lds(g0, g1, z4, z4, z8, 0);
}

// A fragment (16x32 bf16, row-major source, row stride ld), src -> (m=0,k=kbase)
// lanes 0-15: M=lane, K = {0..7, 16..23}; lanes 16-31: M=lane-16, K = {8..15, 24..31}
__device__ __forceinline__ v16bf load_a_bf16(const __bf16* __restrict__ src, int ld, int lane) {
  int mr = lane & 15;
  int kh = (lane >> 4) << 3;           // 0 or 8
  const __bf16* p = src + mr * ld + kh;
  v16bf a;
#pragma unroll
  for (int j = 0; j < 8; ++j) { a[j] = p[j]; a[8 + j] = p[16 + j]; }
  return a;
}

__device__ __forceinline__ v16bf load_a_f32(const float* __restrict__ src, int ld, int lane) {
  int mr = lane & 15;
  int kh = (lane >> 4) << 3;
  const float* p = src + mr * ld + kh;
  v16bf a;
#pragma unroll
  for (int j = 0; j < 8; ++j) { a[j] = (__bf16)p[j]; a[8 + j] = (__bf16)p[16 + j]; }
  return a;
}

// B fragment (32x16 bf16) from transposed layout Bt[n][k] (row stride ld),
// srcT -> (n=0, k=kbase). lanes 0-15: N=lane, K = {0..7, 16..23}; lanes 16-31: N=lane-16, K = {8..15, 24..31}
__device__ __forceinline__ v16bf load_b_bf16(const __bf16* srcT, int ld, int lane) {
  int n  = lane & 15;
  int kh = (lane >> 4) << 3;           // 0 or 8
  const __bf16* p = srcT + n * ld + kh;
  v16bf b;
#pragma unroll
  for (int j = 0; j < 8; ++j) { b[j] = p[j]; b[8 + j] = p[16 + j]; }
  return b;
}

// ---------------------------------------------------------------------------
// Table / weight preparation
// ---------------------------------------------------------------------------

// 16384 threads; each writes one slot of every table group.
__global__ void build_tables(__bf16* __restrict__ TA_c,  __bf16* __restrict__ TA_ms,
                             __bf16* __restrict__ TB_c,  __bf16* __restrict__ TB_sp,
                             __bf16* __restrict__ TB_sm,
                             __bf16* __restrict__ TG_c,  __bf16* __restrict__ TG_sp,
                             __bf16* __restrict__ TG_sm,
                             __bf16* __restrict__ TD_c,  __bf16* __restrict__ TD_ms) {
  int idx = blockIdx.x * blockDim.x + threadIdx.x;
  if (idx >= KM * Ww) return;
  const float TWO_PI_OVER_N = 6.28318530717958647692f / 256.0f;

  { // TA: [ky][w], forward row DFT, e^{-i t}: cos, -sin
    int ky = idx >> 8, w = idx & 255;
    int p = (w * (ky + KY0)) & 255;
    float t = TWO_PI_OVER_N * (float)p;
    TA_c [idx] = (__bf16)__cosf(t);
    TA_ms[idx] = (__bf16)(-__sinf(t));
  }
  { // TB: [kx][h], forward col DFT, e^{-i t}: cos, +sin, -sin variants
    int kx = idx >> 8, h = idx & 255;
    int p = (h * (kx + KX0)) & 255;
    float t = TWO_PI_OVER_N * (float)p;
    float c = __cosf(t), s = __sinf(t);
    TB_c [idx] = (__bf16)c;
    TB_sp[idx] = (__bf16)s;
    TB_sm[idx] = (__bf16)(-s);
  }
  { // TG: [h][kx], inverse col DFT, e^{+i t}
    int h = idx >> 6, kxi = idx & 63;
    int p = (h * (kxi + KX0)) & 255;
    float t = TWO_PI_OVER_N * (float)p;
    float c = __cosf(t), s = __sinf(t);
    TG_c [idx] = (__bf16)c;
    TG_sp[idx] = (__bf16)s;
    TG_sm[idx] = (__bf16)(-s);
  }
  { // TD: [w][ky], inverse row DFT with Hermitian doubling + 1/(H*W)
    int w = idx >> 6, kyi = idx & 63;
    int p = (w * (kyi + KY0)) & 255;
    float t = TWO_PI_OVER_N * (float)p;
    const float sc = 2.0f / 65536.0f;
    TD_c [idx] = (__bf16)(sc * __cosf(t));
    TD_ms[idx] = (__bf16)(-sc * __sinf(t));
  }
}

// Dequantize + transpose weights: q[i][o][mx][my] -> W{re,im}[mode][o][i], bf16
__global__ void dequant_weights(const int* __restrict__ qr, const int* __restrict__ qi,
                                const float* __restrict__ s_r, const float* __restrict__ m_r,
                                const float* __restrict__ s_i, const float* __restrict__ m_i,
                                __bf16* __restrict__ Wre, __bf16* __restrict__ Wim) {
  int idx = blockIdx.x * blockDim.x + threadIdx.x;
  if (idx >= CIN * COUT * KM * KM) return;
  int my = idx & 63;
  int mx = (idx >> 6) & 63;
  int o  = (idx >> 12) & 63;
  int i  = idx >> 18;
  float wr = ((float)qr[idx] + 127.0f) * s_r[0] + m_r[0];
  float wi = ((float)qi[idx] + 127.0f) * s_i[0] + m_i[0];
  int mode = (mx << 6) | my;
  size_t dst = (((size_t)mode * COUT) + o) * CIN + i;
  Wre[dst] = (__bf16)wr;
  Wim[dst] = (__bf16)wi;
}

// ---------------------------------------------------------------------------
// Stage A: pruned row rDFT.  Y[b,ci][ky][h] = sum_w x[h,w] e^{-2pi i w ky/256}
// GEMM: M=h tile(16), K=w(256), N=ky tile(16).  Output transposed to [ky][h].
// Twiddle tables staged into LDS once per block via TDM.
// ---------------------------------------------------------------------------
__global__ __launch_bounds__(256) void stageA(const float* __restrict__ x,
                                              const __bf16* __restrict__ TA_c,
                                              const __bf16* __restrict__ TA_ms,
                                              __bf16* __restrict__ S1re,
                                              __bf16* __restrict__ S1im) {
  __shared__ __bf16 sC [KM * Ww];   // 32 KB
  __shared__ __bf16 sMS[KM * Ww];   // 32 KB
  if ((threadIdx.x >> 5) == 0) {
    tdm_load_1d(lds_addr_of(sC),  TA_c,  (KM * Ww * 2) / 8);
    tdm_load_1d(lds_addr_of(sMS), TA_ms, (KM * Ww * 2) / 8);
    __builtin_amdgcn_s_wait_tensorcnt(0);
  }
  __syncthreads();

  int lane = threadIdx.x & 31;
  int t = blockIdx.x * 8 + (threadIdx.x >> 5);     // 65536 tiles
  int img = t >> 6;                                 // b*64+ci
  int rem = t & 63;
  int h0  = (rem >> 2) << 4;
  int ky0 = (rem & 3) << 4;

  const float* xa = x + (size_t)img * (Hh * Ww) + (size_t)h0 * Ww;
  v8f cre = {}, cim = {};
#pragma unroll
  for (int kk = 0; kk < 8; ++kk) {
    int kb = kk * 32;
    v16bf a  = load_a_f32 (xa + kb, Ww, lane);
    v16bf bc = load_b_bf16(sC  + (size_t)ky0 * Ww + kb, Ww, lane);
    v16bf bm = load_b_bf16(sMS + (size_t)ky0 * Ww + kb, Ww, lane);
    cre = wmma_bf16(a, bc, cre);
    cim = wmma_bf16(a, bm, cim);
  }
  int n = lane & 15, khc = lane >> 4;
  v8bf tr, ti;
#pragma unroll
  for (int r = 0; r < 8; ++r) { tr[r] = (__bf16)cre[r]; ti[r] = (__bf16)cim[r]; }
  size_t base = (size_t)img * (KM * Hh) + (size_t)(ky0 + n) * Hh + h0 + khc * 8;
  *reinterpret_cast<v8bf*>(S1re + base) = tr;
  *reinterpret_cast<v8bf*>(S1im + base) = ti;
}

// ---------------------------------------------------------------------------
// Stage B: pruned col DFT. Z[kx,ky] = sum_h e^{-2pi i kx h/256} Y[h,ky]
// GEMM per image: M=kx(64), K=h(256), N=ky(64), complex.
// Output layout [mode][b][ci] (+ negated imag copy for stage C's A side).
// ---------------------------------------------------------------------------
__global__ __launch_bounds__(256) void stageB(const __bf16* __restrict__ S1re,
                                              const __bf16* __restrict__ S1im,
                                              const __bf16* __restrict__ TB_c,
                                              const __bf16* __restrict__ TB_sp,
                                              const __bf16* __restrict__ TB_sm,
                                              __bf16* __restrict__ S2re,
                                              __bf16* __restrict__ S2im,
                                              __bf16* __restrict__ S2imn) {
  int lane = threadIdx.x & 31;
  int t = blockIdx.x * 8 + (threadIdx.x >> 5);     // 16384 tiles
  int img = t >> 4;
  int kx0 = ((t >> 2) & 3) << 4;
  int ky0 = (t & 3) << 4;

  const __bf16* yre = S1re + (size_t)img * (KM * Hh);
  const __bf16* yim = S1im + (size_t)img * (KM * Hh);
  v8f zre = {}, zim = {};
#pragma unroll
  for (int kk = 0; kk < 8; ++kk) {
    int kb = kk * 32;
    v16bf ac   = load_a_bf16(TB_c  + (size_t)kx0 * Hh + kb, Hh, lane);
    v16bf asp  = load_a_bf16(TB_sp + (size_t)kx0 * Hh + kb, Hh, lane);
    v16bf asm_ = load_a_bf16(TB_sm + (size_t)kx0 * Hh + kb, Hh, lane);
    v16bf bre  = load_b_bf16(yre + (size_t)ky0 * Hh + kb, Hh, lane);
    v16bf bim  = load_b_bf16(yim + (size_t)ky0 * Hh + kb, Hh, lane);
    zre = wmma_bf16(ac,   bre, zre);   // cos*Yre
    zre = wmma_bf16(asp,  bim, zre);   // + sin*Yim
    zim = wmma_bf16(ac,   bim, zim);   // cos*Yim
    zim = wmma_bf16(asm_, bre, zim);   // - sin*Yre
  }
  int n = lane & 15, khc = lane >> 4;
  int b = img >> 6, ci = img & 63;
#pragma unroll
  for (int r = 0; r < 8; ++r) {
    int kx = kx0 + khc * 8 + r;
    int ky = ky0 + n;
    size_t dst = ((size_t)(kx * 64 + ky) * (Bn * CIN)) + (size_t)b * CIN + ci;
    float vr = zre[r], vi = zim[r];
    S2re [dst] = (__bf16)vr;
    S2im [dst] = (__bf16)vi;
    S2imn[dst] = (__bf16)(-vi);
  }
}

// ---------------------------------------------------------------------------
// Stage C: per-mode complex channel mix O[b,o] = sum_i X[b,i] * W[i,o]
// GEMM per mode: M=b(16), K=i(64), N=o(64).  Output [b][o][ky][kx].
// ---------------------------------------------------------------------------
__global__ __launch_bounds__(256) void stageC(const __bf16* __restrict__ S2re,
                                              const __bf16* __restrict__ S2im,
                                              const __bf16* __restrict__ S2imn,
                                              const __bf16* __restrict__ Wre,
                                              const __bf16* __restrict__ Wim,
                                              __bf16* __restrict__ S3re,
                                              __bf16* __restrict__ S3im) {
  int lane = threadIdx.x & 31;
  int t = blockIdx.x * 8 + (threadIdx.x >> 5);     // 16384 tiles
  int mode = t >> 2;
  int o0 = (t & 3) << 4;

  const __bf16* xr  = S2re  + (size_t)mode * (Bn * CIN);
  const __bf16* xi  = S2im  + (size_t)mode * (Bn * CIN);
  const __bf16* xin = S2imn + (size_t)mode * (Bn * CIN);
  const __bf16* wr  = Wre + (size_t)mode * (COUT * CIN) + (size_t)o0 * CIN;
  const __bf16* wi  = Wim + (size_t)mode * (COUT * CIN) + (size_t)o0 * CIN;

  v8f ore = {}, oim = {};
#pragma unroll
  for (int kk = 0; kk < 2; ++kk) {
    int kb = kk * 32;
    v16bf axr  = load_a_bf16(xr  + kb, CIN, lane);
    v16bf axi  = load_a_bf16(xi  + kb, CIN, lane);
    v16bf axin = load_a_bf16(xin + kb, CIN, lane);
    v16bf bwr  = load_b_bf16(wr  + kb, CIN, lane);
    v16bf bwi  = load_b_bf16(wi  + kb, CIN, lane);
    ore = wmma_bf16(axr,  bwr, ore);   // Xre*Wre
    ore = wmma_bf16(axin, bwi, ore);   // - Xim*Wim
    oim = wmma_bf16(axr,  bwi, oim);   // Xre*Wim
    oim = wmma_bf16(axi,  bwr, oim);   // + Xim*Wre
  }
  int n = lane & 15, khc = lane >> 4;
  int kx = mode >> 6, ky = mode & 63;
#pragma unroll
  for (int r = 0; r < 8; ++r) {
    int b = khc * 8 + r;
    int o = o0 + n;
    size_t dst = (((size_t)b * COUT + o) * KM + ky) * KM + kx;
    S3re[dst] = (__bf16)ore[r];
    S3im[dst] = (__bf16)oim[r];
  }
}

// ---------------------------------------------------------------------------
// Stage D1: inverse col DFT. P[h,ky] = sum_kx O[kx,ky] e^{+2pi i kx h/256}
// GEMM per image(b,o): M=h(256), K=kx(64), N=ky(64).  Output [img][h][ky].
// ---------------------------------------------------------------------------
__global__ __launch_bounds__(256) void stageD1(const __bf16* __restrict__ S3re,
                                               const __bf16* __restrict__ S3im,
                                               const __bf16* __restrict__ TG_c,
                                               const __bf16* __restrict__ TG_sp,
                                               const __bf16* __restrict__ TG_sm,
                                               __bf16* __restrict__ S4re,
                                               __bf16* __restrict__ S4im) {
  int lane = threadIdx.x & 31;
  int t = blockIdx.x * 8 + (threadIdx.x >> 5);     // 65536 tiles
  int img = t >> 6;                                 // b*64+o
  int rem = t & 63;
  int h0  = (rem >> 2) << 4;
  int ky0 = (rem & 3) << 4;

  const __bf16* ore = S3re + (size_t)img * (KM * KM);
  const __bf16* oim = S3im + (size_t)img * (KM * KM);
  v8f pre = {}, pim = {};
#pragma unroll
  for (int kk = 0; kk < 2; ++kk) {
    int kb = kk * 32;
    v16bf ac   = load_a_bf16(TG_c  + (size_t)h0 * KM + kb, KM, lane);
    v16bf asp  = load_a_bf16(TG_sp + (size_t)h0 * KM + kb, KM, lane);
    v16bf asm_ = load_a_bf16(TG_sm + (size_t)h0 * KM + kb, KM, lane);
    v16bf bor  = load_b_bf16(ore + (size_t)ky0 * KM + kb, KM, lane);
    v16bf boi  = load_b_bf16(oim + (size_t)ky0 * KM + kb, KM, lane);
    pre = wmma_bf16(ac,   bor, pre);   // cos*Ore
    pre = wmma_bf16(asm_, boi, pre);   // - sin*Oim
    pim = wmma_bf16(ac,   boi, pim);   // cos*Oim
    pim = wmma_bf16(asp,  bor, pim);   // + sin*Ore
  }
  int n = lane & 15, khc = lane >> 4;
#pragma unroll
  for (int r = 0; r < 8; ++r) {
    int h = h0 + khc * 8 + r;
    size_t dst = (size_t)img * (Hh * KM) + (size_t)h * KM + (ky0 + n);
    S4re[dst] = (__bf16)pre[r];
    S4im[dst] = (__bf16)pim[r];
  }
}

// ---------------------------------------------------------------------------
// Stage D2: inverse row rDFT (Hermitian) + bias -> fp32 output.
// out[h,w] = sum_ky Pre*c2[ky,w] + Pim*s2m[ky,w], tables carry 2/(H*W).
// GEMM per image: M=h(256), K=ky(64), N=w(256).  TD tables staged via TDM.
// ---------------------------------------------------------------------------
__global__ __launch_bounds__(256) void stageD2(const __bf16* __restrict__ S4re,
                                               const __bf16* __restrict__ S4im,
                                               const __bf16* __restrict__ TD_c,
                                               const __bf16* __restrict__ TD_ms,
                                               const int* __restrict__ qbias,
                                               const float* __restrict__ b_scale,
                                               const float* __restrict__ b_min,
                                               float* __restrict__ out) {
  __shared__ __bf16 sC [Ww * KM];   // 32 KB  [w][ky]
  __shared__ __bf16 sMS[Ww * KM];   // 32 KB
  if ((threadIdx.x >> 5) == 0) {
    tdm_load_1d(lds_addr_of(sC),  TD_c,  (Ww * KM * 2) / 8);
    tdm_load_1d(lds_addr_of(sMS), TD_ms, (Ww * KM * 2) / 8);
    __builtin_amdgcn_s_wait_tensorcnt(0);
  }
  __syncthreads();

  int lane = threadIdx.x & 31;
  int t = blockIdx.x * 8 + (threadIdx.x >> 5);     // 262144 tiles
  int img = t >> 8;                                 // b*64+o
  int rem = t & 255;
  int h0 = (rem >> 4) << 4;
  int w0 = (rem & 15) << 4;

  const __bf16* pre = S4re + (size_t)img * (Hh * KM);
  const __bf16* pim = S4im + (size_t)img * (Hh * KM);
  v8f acc = {};
#pragma unroll
  for (int kk = 0; kk < 2; ++kk) {
    int kb = kk * 32;
    v16bf are = load_a_bf16(pre + (size_t)h0 * KM + kb, KM, lane);
    v16bf aim = load_a_bf16(pim + (size_t)h0 * KM + kb, KM, lane);
    v16bf bc  = load_b_bf16(sC  + (size_t)w0 * KM + kb, KM, lane);
    v16bf bm  = load_b_bf16(sMS + (size_t)w0 * KM + kb, KM, lane);
    acc = wmma_bf16(are, bc, acc);
    acc = wmma_bf16(aim, bm, acc);
  }
  int o = img & 63;
  float bias = ((float)qbias[o] + 127.0f) * b_scale[0] + b_min[0];
  int n = lane & 15, khc = lane >> 4;
#pragma unroll
  for (int r = 0; r < 8; ++r) {
    int h = h0 + khc * 8 + r;
    out[(size_t)img * (Hh * Ww) + (size_t)h * Ww + (w0 + n)] = acc[r] + bias;
  }
}

// ---------------------------------------------------------------------------
// Host launcher
// ---------------------------------------------------------------------------
extern "C" void kernel_launch(void* const* d_in, const int* in_sizes, int n_in,
                              void* d_out, int out_size, void* d_ws, size_t ws_size,
                              hipStream_t stream) {
  const float* x      = (const float*)d_in[0];
  const int*   q_real = (const int*)  d_in[1];
  const int*   q_imag = (const int*)  d_in[2];
  const int*   q_bias = (const int*)  d_in[3];
  const float* sc_r   = (const float*)d_in[4];
  const float* mn_r   = (const float*)d_in[5];
  const float* sc_i   = (const float*)d_in[6];
  const float* mn_i   = (const float*)d_in[7];
  const float* b_sc   = (const float*)d_in[8];
  const float* b_mn   = (const float*)d_in[9];
  float* out = (float*)d_out;

  // workspace carve-up
  char* p = (char*)d_ws;
  auto alloc = [&](size_t bytes) -> char* {
    char* r = p;
    p += (bytes + 255) & ~(size_t)255;
    return r;
  };
  const size_t TBL = (size_t)KM * 256 * sizeof(__bf16);         // 32 KB each
  __bf16* TA_c  = (__bf16*)alloc(TBL);
  __bf16* TA_ms = (__bf16*)alloc(TBL);
  __bf16* TB_c  = (__bf16*)alloc(TBL);
  __bf16* TB_sp = (__bf16*)alloc(TBL);
  __bf16* TB_sm = (__bf16*)alloc(TBL);
  __bf16* TG_c  = (__bf16*)alloc(TBL);
  __bf16* TG_sp = (__bf16*)alloc(TBL);
  __bf16* TG_sm = (__bf16*)alloc(TBL);
  __bf16* TD_c  = (__bf16*)alloc(TBL);
  __bf16* TD_ms = (__bf16*)alloc(TBL);

  const size_t WSZ  = (size_t)KM * KM * COUT * CIN * sizeof(__bf16);   // 33.5 MB
  __bf16* Wre = (__bf16*)alloc(WSZ);
  __bf16* Wim = (__bf16*)alloc(WSZ);

  const size_t S1SZ = (size_t)Bn * CIN * KM * Hh * sizeof(__bf16);     // 33.5 MB
  __bf16* S1re = (__bf16*)alloc(S1SZ);
  __bf16* S1im = (__bf16*)alloc(S1SZ);

  const size_t S2SZ = (size_t)KM * KM * Bn * CIN * sizeof(__bf16);     //  8.4 MB
  __bf16* S2re  = (__bf16*)alloc(S2SZ);
  __bf16* S2im  = (__bf16*)alloc(S2SZ);
  __bf16* S2imn = (__bf16*)alloc(S2SZ);

  const size_t S3SZ = (size_t)Bn * COUT * KM * KM * sizeof(__bf16);    //  8.4 MB
  __bf16* S3re = (__bf16*)alloc(S3SZ);
  __bf16* S3im = (__bf16*)alloc(S3SZ);

  // S4 aliases S1 (S1 is dead after stage B completes; stream-ordered)
  __bf16* S4re = S1re;
  __bf16* S4im = S1im;

  if ((size_t)(p - (char*)d_ws) > ws_size) return;  // ws too small: bail deterministically

  // prep
  build_tables<<<64, 256, 0, stream>>>(TA_c, TA_ms, TB_c, TB_sp, TB_sm,
                                       TG_c, TG_sp, TG_sm, TD_c, TD_ms);
  dequant_weights<<<(CIN * COUT * KM * KM) / 256, 256, 0, stream>>>(
      q_real, q_imag, sc_r, mn_r, sc_i, mn_i, Wre, Wim);

  // pipeline (one wave per 16x16 WMMA tile, 8 waves / 256-thread block)
  stageA <<< 65536 / 8, 256, 0, stream>>>(x, TA_c, TA_ms, S1re, S1im);
  stageB <<< 16384 / 8, 256, 0, stream>>>(S1re, S1im, TB_c, TB_sp, TB_sm,
                                          S2re, S2im, S2imn);
  stageC <<< 16384 / 8, 256, 0, stream>>>(S2re, S2im, S2imn, Wre, Wim, S3re, S3im);
  stageD1<<< 65536 / 8, 256, 0, stream>>>(S3re, S3im, TG_c, TG_sp, TG_sm, S4re, S4im);
  stageD2<<<262144 / 8, 256, 0, stream>>>(S4re, S4im, TD_c, TD_ms,
                                          q_bias, b_sc, b_mn, out);
}
